// mnist_RNN_model_39840116638205
// MI455X (gfx1250) — matrix operand order repaired and checked by reading, not compiled, not run on hardware
//
#include <hip/hip_runtime.h>
#include <hip/hip_bf16.h>
#include <math.h>

// ---------- types ----------
typedef __attribute__((ext_vector_type(16))) __bf16 v16bf;
typedef __attribute__((ext_vector_type(8)))  __bf16 bf16x8;
typedef __attribute__((ext_vector_type(8)))  float  v8f;

#define BDIM 4096
#define HDIM 512
#define TDIM 28
#define IDIM 28
#define IPAD 32     // K-padded input width
#define GDIM 1536   // 3*H
#define G2   2048   // gate buffer row: [r|z|gx_n|gh_n]

// ------------------------------------------------------------------
// Fragment loaders (layouts per CDNA5 ISA 7.12.2, wave32).
// All K offsets are multiples of 32 elements -> no tail guards.
// A: 16x32 bf16. lane = kg*16 + m ; j<8 -> k=kg*8+j ; j>=8 -> k=16+kg*8+(j-8)
// ------------------------------------------------------------------
__device__ inline v16bf load_a_frag(const __bf16* __restrict__ A, int lda,
                                    int row0, int k0) {
  int lane = threadIdx.x & 31;
  int mr = lane & 15, kg = lane >> 4;
  const __bf16* rowp = A + (size_t)(row0 + mr) * (size_t)lda;
  bf16x8 lo = *(const bf16x8*)(rowp + k0 + kg * 8);        // 16B aligned
  bf16x8 hi = *(const bf16x8*)(rowp + k0 + 16 + kg * 8);   // 16B aligned
  v16bf f;
#pragma unroll
  for (int i = 0; i < 8; ++i) { f[i] = lo[i]; f[8 + i] = hi[i]; }
  return f;
}

// B: 32x16 bf16 = W^T tile, W stored N x K row-major (bf16).
// lane = kg*16 + n ; j<8 -> k = k0+kg*8+j ; j>=8 -> k = k0+16+kg*8+(j-8)
__device__ inline v16bf load_b_frag(const __bf16* __restrict__ W, int ldw,
                                    int n0, int k0) {
  int lane = threadIdx.x & 31;
  int nn = lane & 15, kg = lane >> 4;
  const __bf16* rowp = W + (size_t)(n0 + nn) * (size_t)ldw;
  bf16x8 lo = *(const bf16x8*)(rowp + k0 + kg * 8);
  bf16x8 hi = *(const bf16x8*)(rowp + k0 + 16 + kg * 8);
  v16bf f;
#pragma unroll
  for (int i = 0; i < 8; ++i) { f[i] = lo[i]; f[8 + i] = hi[i]; }
  return f;
}

// C/D 16x16 f32: lane = rh*16 + n, VGPR v -> m = rh*8 + v
__device__ inline void store_tiles(float* __restrict__ out, int ldo,
                                   int m0, int n0, int coloff, const v8f* acc,
                                   const float* b1, const float* b2,
                                   const float* scale, const float* shift) {
  int lane = threadIdx.x & 31;
  int nn = lane & 15, rh = lane >> 4;
#pragma unroll
  for (int mm = 0; mm < 2; ++mm) {
#pragma unroll
    for (int j = 0; j < 4; ++j) {
      int col = n0 + j * 16 + nn;
      float bias = (b1 ? b1[col] : 0.0f) + (b2 ? b2[col] : 0.0f);
      v8f a = acc[mm * 4 + j];
#pragma unroll
      for (int v = 0; v < 8; ++v) {
        int row = m0 + mm * 16 + rh * 8 + v;
        float val = a[v] + bias;
        if (scale) { val = val * scale[col] + shift[col]; val = val > 0.0f ? val : 0.0f; }
        out[(size_t)row * (size_t)ldo + col + coloff] = val;
      }
    }
  }
}

// ------------------------------------------------------------------
// Generic two-phase WMMA GEMM: out = A1@W1^T (+ A2@W2^T) + bias
// Block: 256 thr (8 waves), block tile 64(M) x 256(N); wave tile 32x64.
// Column strips with n0 >= split_col keep phase-1 and phase-2 results
// separate (GRU n-gate): phase1 -> col n, phase2 -> col n+512.
// ------------------------------------------------------------------
__global__ __launch_bounds__(256) void wmma_gemm(
    const __bf16* __restrict__ A1, int lda1, int K1,
    const __bf16* __restrict__ A2, int lda2, int K2,
    const __bf16* __restrict__ W1, const __bf16* __restrict__ W2,
    const float* __restrict__ b1, const float* __restrict__ b2,
    float* __restrict__ out, int ldo, int split_col,
    const float* __restrict__ bn_scale, const float* __restrict__ bn_shift) {
  int wave = threadIdx.x >> 5;
  int wm = wave & 1, wn = wave >> 1;
  int m0 = blockIdx.x * 64 + wm * 32;
  int n0 = blockIdx.y * 256 + wn * 64;
  bool split = (n0 >= split_col);

  v8f acc[8];
#pragma unroll
  for (int i = 0; i < 8; ++i)
#pragma unroll
    for (int e = 0; e < 8; ++e) acc[i][e] = 0.0f;

  // phase 1: A1 (M x K1) x W1^T
  for (int k0 = 0; k0 < K1; k0 += 32) {
    v16bf a0 = load_a_frag(A1, lda1, m0, k0);
    v16bf a1 = load_a_frag(A1, lda1, m0 + 16, k0);
#pragma unroll
    for (int j = 0; j < 4; ++j) {
      v16bf b = load_b_frag(W1, K1, n0 + j * 16, k0);
      acc[j]     = __builtin_amdgcn_wmma_f32_16x16x32_bf16(false, a0, false, b, (short)0, acc[j],     false, false);
      acc[4 + j] = __builtin_amdgcn_wmma_f32_16x16x32_bf16(false, a1, false, b, (short)0, acc[4 + j], false, false);
    }
  }

  if (split) {                       // dump gx_n (+b_ih) then restart for gh_n
    store_tiles(out, ldo, m0, n0, 0, acc, b1, nullptr, nullptr, nullptr);
#pragma unroll
    for (int i = 0; i < 8; ++i)
#pragma unroll
      for (int e = 0; e < 8; ++e) acc[i][e] = 0.0f;
  }

  // phase 2: A2 (M x K2) x W2^T
  for (int k0 = 0; k0 < K2; k0 += 32) {
    v16bf a0 = load_a_frag(A2, lda2, m0, k0);
    v16bf a1 = load_a_frag(A2, lda2, m0 + 16, k0);
#pragma unroll
    for (int j = 0; j < 4; ++j) {
      v16bf b = load_b_frag(W2, K2, n0 + j * 16, k0);
      acc[j]     = __builtin_amdgcn_wmma_f32_16x16x32_bf16(false, a0, false, b, (short)0, acc[j],     false, false);
      acc[4 + j] = __builtin_amdgcn_wmma_f32_16x16x32_bf16(false, a1, false, b, (short)0, acc[4 + j], false, false);
    }
  }

  if (split)
    store_tiles(out, ldo, m0, n0, 512, acc, b2, nullptr, nullptr, nullptr);
  else
    store_tiles(out, ldo, m0, n0, 0, acc, b1, b2, bn_scale, bn_shift);
}

// ------------------------------------------------------------------
// GRU gate update. g row: [r-sum | z-sum | gx_n | gh_n] (each H wide)
// Writes fp32 recurrent state h, bf16 copy hb (next GEMM's A), and
// optionally the bf16 sequence output y.
// ------------------------------------------------------------------
__global__ __launch_bounds__(256) void gru_gate_kernel(
    const float* __restrict__ g, float* __restrict__ h, __bf16* __restrict__ hb,
    __bf16* __restrict__ y, int ldy, int ycol) {
  int idx = blockIdx.x * 256 + threadIdx.x;   // B*H threads
  int b = idx >> 9, j = idx & 511;
  const float* gr = g + (size_t)b * G2;
  float r = 1.0f / (1.0f + __expf(-gr[j]));
  float z = 1.0f / (1.0f + __expf(-gr[512 + j]));
  float n = tanhf(gr[1024 + j] + r * gr[1536 + j]);
  float ho = h[idx];
  float hn = (1.0f - z) * n + z * ho;
  h[idx] = hn;
  hb[idx] = (__bf16)hn;
  if (y) y[(size_t)b * (size_t)ldy + ycol + j] = (__bf16)hn;
}

// ---------- conversion / init helpers ----------
__global__ void zero_f32(float* __restrict__ p, int n) {
  int i = blockIdx.x * 256 + threadIdx.x;
  if (i < n) p[i] = 0.0f;
}
__global__ void zero_bf16(__bf16* __restrict__ p, int n) {
  int i = blockIdx.x * 256 + threadIdx.x;
  if (i < n) p[i] = (__bf16)0.0f;
}
__global__ void conv_bf16(const float* __restrict__ src, __bf16* __restrict__ dst, int n) {
  int i = blockIdx.x * 256 + threadIdx.x;
  if (i < n) dst[i] = (__bf16)src[i];
}
// pad row length Ks -> Kd with zeros while converting
__global__ void pad_conv_bf16(const float* __restrict__ src, __bf16* __restrict__ dst,
                              int rows, int Ks, int Kd) {
  int i = blockIdx.x * 256 + threadIdx.x;
  if (i >= rows * Kd) return;
  int r = i / Kd, c = i - r * Kd;
  dst[i] = (c < Ks) ? (__bf16)src[(size_t)r * Ks + c] : (__bf16)0.0f;
}

__global__ void bn_prep(const float* __restrict__ gamma, const float* __restrict__ beta,
                        const float* __restrict__ mean, const float* __restrict__ var,
                        float* __restrict__ scale, float* __restrict__ shift) {
  int i = threadIdx.x;   // 256 channels
  float s = gamma[i] * rsqrtf(var[i] + 1e-5f);
  scale[i] = s;
  shift[i] = beta[i] - mean[i] * s;
}

// fc2 (256->10) + log_softmax; one block of 256 threads per row
__global__ __launch_bounds__(256) void fc2_logsoftmax(
    const float* __restrict__ act, const float* __restrict__ w,
    const float* __restrict__ bias, float* __restrict__ out) {
  __shared__ float red[256];
  __shared__ float logits[16];
  int row = blockIdx.x, t = threadIdx.x;
  float a = act[(size_t)row * 256 + t];
  for (int n = 0; n < 10; ++n) {
    red[t] = a * w[n * 256 + t];
    __syncthreads();
    for (int s = 128; s > 0; s >>= 1) {
      if (t < s) red[t] += red[t + s];
      __syncthreads();
    }
    if (t == 0) logits[n] = red[0] + bias[n];
    __syncthreads();
  }
  if (t == 0) {
    float m = logits[0];
    for (int n = 1; n < 10; ++n) m = fmaxf(m, logits[n]);
    float s = 0.0f;
    for (int n = 0; n < 10; ++n) s += __expf(logits[n] - m);
    float lse = m + __logf(s);
    for (int n = 0; n < 10; ++n) out[(size_t)row * 10 + n] = logits[n] - lse;
  }
}

// ------------------------------------------------------------------
extern "C" void kernel_launch(void* const* d_in, const int* in_sizes, int n_in,
                              void* d_out, int out_size, void* d_ws, size_t ws_size,
                              hipStream_t stream) {
  const float* x        = (const float*)d_in[0];
  const float* W_ih_l0  = (const float*)d_in[1];
  const float* W_hh_l0  = (const float*)d_in[2];
  const float* b_ih_l0  = (const float*)d_in[3];
  const float* b_hh_l0  = (const float*)d_in[4];
  const float* W_ih_l0r = (const float*)d_in[5];
  const float* W_hh_l0r = (const float*)d_in[6];
  const float* b_ih_l0r = (const float*)d_in[7];
  const float* b_hh_l0r = (const float*)d_in[8];
  const float* W_ih_l1  = (const float*)d_in[9];
  const float* W_hh_l1  = (const float*)d_in[10];
  const float* b_ih_l1  = (const float*)d_in[11];
  const float* b_hh_l1  = (const float*)d_in[12];
  const float* W_ih_l1r = (const float*)d_in[13];
  const float* W_hh_l1r = (const float*)d_in[14];
  const float* b_ih_l1r = (const float*)d_in[15];
  const float* b_hh_l1r = (const float*)d_in[16];
  const float* fc1_w    = (const float*)d_in[17];
  const float* fc1_b    = (const float*)d_in[18];
  const float* bn_gamma = (const float*)d_in[19];
  const float* bn_beta  = (const float*)d_in[20];
  const float* bn_mean  = (const float*)d_in[21];
  const float* bn_var   = (const float*)d_in[22];
  const float* fc2_w    = (const float*)d_in[23];
  const float* fc2_b    = (const float*)d_in[24];
  float* out = (float*)d_out;

  // ---------------- workspace layout ----------------
  // fp32 region first, then bf16 region (all sizes keep 32B alignment)
  float* wsf = (float*)d_ws;
  size_t off = 0;
  float* goutF = wsf + off; off += (size_t)BDIM * G2;
  float* goutB = wsf + off; off += (size_t)BDIM * G2;
  float* h0f   = wsf + off; off += (size_t)BDIM * HDIM;
  float* h0b   = wsf + off; off += (size_t)BDIM * HDIM;
  float* h1f   = wsf + off; off += (size_t)BDIM * HDIM;
  float* h1b   = wsf + off; off += (size_t)BDIM * HDIM;
  float* act   = wsf + off; off += (size_t)BDIM * 256;
  float* bnsc  = wsf + off; off += 256;
  float* bnsh  = wsf + off; off += 256;

  __bf16* wsb = (__bf16*)(wsf + off);
  size_t bo = 0;
  __bf16* xpad  = wsb + bo; bo += (size_t)BDIM * TDIM * IPAD;   // padded bf16 x
  __bf16* y0b   = wsb + bo; bo += (size_t)BDIM * TDIM * 1024;   // layer0 output (bf16)
  __bf16* hb0f  = wsb + bo; bo += (size_t)BDIM * HDIM;
  __bf16* hb0b  = wsb + bo; bo += (size_t)BDIM * HDIM;
  __bf16* hb1f  = wsb + bo; bo += (size_t)BDIM * HDIM;
  __bf16* hb1b  = wsb + bo; bo += (size_t)BDIM * HDIM;
  __bf16* lastb = wsb + bo; bo += (size_t)BDIM * 1024;
  __bf16* wbih0  = wsb + bo; bo += (size_t)GDIM * IPAD;   // padded 1536x32
  __bf16* wbih0r = wsb + bo; bo += (size_t)GDIM * IPAD;
  __bf16* wbhh0  = wsb + bo; bo += (size_t)GDIM * HDIM;
  __bf16* wbhh0r = wsb + bo; bo += (size_t)GDIM * HDIM;
  __bf16* wbih1  = wsb + bo; bo += (size_t)GDIM * 1024;
  __bf16* wbih1r = wsb + bo; bo += (size_t)GDIM * 1024;
  __bf16* wbhh1  = wsb + bo; bo += (size_t)GDIM * HDIM;
  __bf16* wbhh1r = wsb + bo; bo += (size_t)GDIM * HDIM;
  __bf16* wbfc1  = wsb + bo; bo += (size_t)256 * 1024;

  dim3 blk(256);
  dim3 gridG(BDIM / 64, GDIM / 256);   // GRU GEMMs: 4096 x 1536
  dim3 gridF(BDIM / 64, 1);            // fc1: 4096 x 256
  const int gateBlocks = (BDIM * HDIM) / 256;
  auto nb = [](size_t n) { return (unsigned)((n + 255) / 256); };

  // ---------------- operand conversion (per call; inputs may change) ----
  pad_conv_bf16<<<nb((size_t)BDIM * TDIM * IPAD), blk, 0, stream>>>(x, xpad, BDIM * TDIM, IDIM, IPAD);
  pad_conv_bf16<<<nb((size_t)GDIM * IPAD), blk, 0, stream>>>(W_ih_l0,  wbih0,  GDIM, IDIM, IPAD);
  pad_conv_bf16<<<nb((size_t)GDIM * IPAD), blk, 0, stream>>>(W_ih_l0r, wbih0r, GDIM, IDIM, IPAD);
  conv_bf16<<<nb((size_t)GDIM * HDIM), blk, 0, stream>>>(W_hh_l0,  wbhh0,  GDIM * HDIM);
  conv_bf16<<<nb((size_t)GDIM * HDIM), blk, 0, stream>>>(W_hh_l0r, wbhh0r, GDIM * HDIM);
  conv_bf16<<<nb((size_t)GDIM * 1024), blk, 0, stream>>>(W_ih_l1,  wbih1,  GDIM * 1024);
  conv_bf16<<<nb((size_t)GDIM * 1024), blk, 0, stream>>>(W_ih_l1r, wbih1r, GDIM * 1024);
  conv_bf16<<<nb((size_t)GDIM * HDIM), blk, 0, stream>>>(W_hh_l1,  wbhh1,  GDIM * HDIM);
  conv_bf16<<<nb((size_t)GDIM * HDIM), blk, 0, stream>>>(W_hh_l1r, wbhh1r, GDIM * HDIM);
  conv_bf16<<<nb((size_t)256 * 1024), blk, 0, stream>>>(fc1_w, wbfc1, 256 * 1024);

  // zero recurrent state (fp32 x4 and bf16 x4, each contiguous)
  zero_f32 <<<nb(4 * BDIM * HDIM), blk, 0, stream>>>(h0f,  4 * BDIM * HDIM);
  zero_bf16<<<nb(4 * BDIM * HDIM), blk, 0, stream>>>(hb0f, 4 * BDIM * HDIM);

  // ---------------- layer 0 (both directions, full T) ----------------
  for (int s = 0; s < TDIM; ++s) {
    wmma_gemm<<<gridG, blk, 0, stream>>>(xpad + s * IPAD, TDIM * IPAD, IPAD,
                                         hb0f, HDIM, HDIM,
                                         wbih0, wbhh0, b_ih_l0, b_hh_l0,
                                         goutF, G2, 1024, nullptr, nullptr);
    gru_gate_kernel<<<gateBlocks, blk, 0, stream>>>(goutF, h0f, hb0f, y0b, TDIM * 1024, s * 1024);

    int tb = TDIM - 1 - s;
    wmma_gemm<<<gridG, blk, 0, stream>>>(xpad + tb * IPAD, TDIM * IPAD, IPAD,
                                         hb0b, HDIM, HDIM,
                                         wbih0r, wbhh0r, b_ih_l0r, b_hh_l0r,
                                         goutB, G2, 1024, nullptr, nullptr);
    gru_gate_kernel<<<gateBlocks, blk, 0, stream>>>(goutB, h0b, hb0b, y0b, TDIM * 1024, tb * 1024 + 512);
  }

  // ---------------- layer 1 forward (full T; only final h needed) ----
  for (int s = 0; s < TDIM; ++s) {
    wmma_gemm<<<gridG, blk, 0, stream>>>(y0b + s * 1024, TDIM * 1024, 1024,
                                         hb1f, HDIM, HDIM,
                                         wbih1, wbhh1, b_ih_l1, b_hh_l1,
                                         goutF, G2, 1024, nullptr, nullptr);
    gru_gate_kernel<<<gateBlocks, blk, 0, stream>>>(
        goutF, h1f, hb1f, (s == TDIM - 1) ? lastb : nullptr, 1024, 0);
  }

  // ---------------- layer 1 backward: only step at t = T-1 -----------
  wmma_gemm<<<gridG, blk, 0, stream>>>(y0b + (TDIM - 1) * 1024, TDIM * 1024, 1024,
                                       hb1b, HDIM, HDIM,
                                       wbih1r, wbhh1r, b_ih_l1r, b_hh_l1r,
                                       goutB, G2, 1024, nullptr, nullptr);
  gru_gate_kernel<<<gateBlocks, blk, 0, stream>>>(goutB, h1b, hb1b, lastb, 1024, 512);

  // ---------------- FC head ------------------------------------------
  bn_prep<<<1, blk, 0, stream>>>(bn_gamma, bn_beta, bn_mean, bn_var, bnsc, bnsh);
  wmma_gemm<<<gridF, blk, 0, stream>>>(lastb, 1024, 1024,
                                       nullptr, 0, 0,
                                       wbfc1, nullptr, fc1_b, nullptr,
                                       act, 256, 1 << 30, bnsc, bnsh);
  fc2_logsoftmax<<<BDIM, blk, 0, stream>>>(act, fc2_w, fc2_b, out);
}
